// EmotionNet_21766894256751
// MI455X (gfx1250) — matrix operand order repaired and checked by reading, not compiled, not run on hardware
//
#include <hip/hip_runtime.h>
#include <hip/hip_bf16.h>

// Seq2seq LSTM for MI455X (gfx1250, wave32, WMMA).
//
// Strategy (compute/latency-bound recurrent GEMM, everything L2-resident):
//  * Batch rows are independent through the recurrence -> 16 persistent
//    workgroups, each owning 16 batch rows (= WMMA M) for all 612 steps.
//    NO inter-workgroup sync ever needed.
//  * h state lives in LDS as bf16 (padded rows, feeds WMMA A fragments via
//    2x ds_load_b128 per k-tile); c state lives in registers (C-tile layout).
//  * Whh pre-converted to bf16 row-major so each lane's B fragment is 16
//    contiguous bf16 (2x global_load_b128, served from L2; 2MB per matrix).
//  * Each WG: 16 wave32s; wave w owns hidden indices [32w,32w+32) => all 4
//    gate column groups for those indices => i/f/g/o tiles share lanes/regs,
//    cell nonlinearity is pure register math, h_new written back to LDS.
//  * v_wmma_f32_16x16x32_bf16, f32 accumulate. 128 WMMA / wave / step.

#define Hdim   512
#define Bdim   256
#define Sdim   512
#define LENdim 100
#define NCdim  3
#define HSTR   520   // padded LDS row stride in bf16 elems (1040B, 16B aligned, bank-spread)

typedef __attribute__((ext_vector_type(16))) __bf16 v16bf;
typedef __attribute__((ext_vector_type(8)))  __bf16 v8bf;
typedef __attribute__((ext_vector_type(8)))  float  v8f;

__device__ __forceinline__ float sigm_f(float x)   { return 1.0f / (1.0f + __expf(-x)); }
__device__ __forceinline__ float tanh_fast(float x){ float e = __expf(-2.0f*x); return (1.0f - e) / (1.0f + e); }

// One-time prep: fp32 Whh -> bf16 (RNE via hardware cvt), bias sums.
__global__ void prep_kernel(const float* __restrict__ encW, const float* __restrict__ decW,
                            unsigned short* __restrict__ encWb, unsigned short* __restrict__ decWb,
                            const float* __restrict__ ebih, const float* __restrict__ ebhh,
                            const float* __restrict__ dbih, const float* __restrict__ dbhh,
                            float* __restrict__ ebias, float* __restrict__ dbias)
{
    int i = blockIdx.x * blockDim.x + threadIdx.x;
    if (i < 4 * Hdim * Hdim) {
        union { __bf16 b; unsigned short u; } ce, cd;
        ce.b = (__bf16)encW[i];
        cd.b = (__bf16)decW[i];
        encWb[i] = ce.u;
        decWb[i] = cd.u;
    }
    if (i < 4 * Hdim) {
        ebias[i] = ebih[i] + ebhh[i];
        dbias[i] = dbih[i] + dbhh[i];
    }
}

__global__ __launch_bounds__(512)
void lstm_seq2seq_kernel(const float* __restrict__ src, const float* __restrict__ tgt,
                         const float* __restrict__ encWih, const float* __restrict__ decWih,
                         const float* __restrict__ fcW, const float* __restrict__ fcB,
                         const unsigned short* __restrict__ encWu,
                         const unsigned short* __restrict__ decWu,
                         const float* __restrict__ ebias, const float* __restrict__ dbias,
                         float* __restrict__ out)
{
    __shared__ __bf16 h_lds[16 * HSTR];      // h state, 16 batch rows x 512 (padded)
    __shared__ float  xlds[16];              // current scalar input per batch row
    __shared__ float  fcw_s[NCdim * Hdim];   // fc weights
    __shared__ float  fcb_s[NCdim];

    const int tid  = threadIdx.x;
    const int lane = tid & 31;
    const int wave = tid >> 5;         // 0..15
    const int l16  = lane & 15;
    const int half = lane >> 4;        // 0/1
    const int b0   = blockIdx.x << 4;  // batch base for this WG
    const int jb   = wave << 5;        // hidden-index base for this wave

    const __bf16* encW = (const __bf16*)encWu;
    const __bf16* decW = (const __bf16*)decWu;

    for (int i = tid; i < 16 * HSTR; i += 512) h_lds[i] = (__bf16)0.0f;
    for (int i = tid; i < NCdim * Hdim; i += 512) fcw_s[i] = fcW[i];
    if (tid < NCdim) fcb_s[tid] = fcB[tid];
    if (tid < 16)    xlds[tid] = src[(b0 + tid) * Sdim];   // x at t=0

    v8f cst[2];                                // c state (C-tile layout), regs only
    #pragma unroll
    for (int tj = 0; tj < 2; ++tj)
        #pragma unroll
        for (int r = 0; r < 8; ++r) cst[tj][r] = 0.0f;

    float wih_r[8], bias_r[8];                 // per-phase gate constants, one per owned tile

    auto load_consts = [&](const float* wih, const float* bias) {
        #pragma unroll
        for (int g = 0; g < 4; ++g)
            #pragma unroll
            for (int tj = 0; tj < 2; ++tj) {
                int col = g * Hdim + jb + tj * 16 + l16;
                wih_r[g * 2 + tj]  = wih[col];
                bias_r[g * 2 + tj] = bias[col];
            }
    };

    auto do_step = [&](const __bf16* __restrict__ W) {
        v8f acc[4][2];
        // Fold x_t * Wih + (bih+bhh) into accumulator init (C layout: m=r+8*half, n=l16)
        #pragma unroll
        for (int g = 0; g < 4; ++g)
            #pragma unroll
            for (int tj = 0; tj < 2; ++tj)
                #pragma unroll
                for (int r = 0; r < 8; ++r)
                    acc[g][tj][r] = xlds[r + 8 * half] * wih_r[g * 2 + tj] + bias_r[g * 2 + tj];

        // GEMM: gates(16 x 128-of-2048) += h(16x512) @ Whh^T slice, K in 16 steps of 32
        const __bf16* hrow = &h_lds[l16 * HSTR];   // A: lane = row M
        #pragma unroll 2
        for (int kt = 0; kt < 16; ++kt) {
            const int k0 = kt * 32;
            // A 16x32 bf16, this code's k order: lane m, VGPR0-3 <- K 16h+0..7, VGPR4-7 <- K 16h+8..15
            v8bf alo = *(const v8bf*)(hrow + k0 + 16 * half);
            v8bf ahi = *(const v8bf*)(hrow + k0 + 16 * half + 8);
            v16bf a;
            #pragma unroll
            for (int i = 0; i < 8; ++i) { a[i] = alo[i]; a[i + 8] = ahi[i]; }
            #pragma unroll
            for (int g = 0; g < 4; ++g)
                #pragma unroll
                for (int tj = 0; tj < 2; ++tj) {
                    // B 32x16 bf16, same k order as A: lane = col N, half selects K 0-15 / 16-31,
                    // 16 contiguous bf16 from row (col) of row-major Whh.
                    const int col = g * Hdim + jb + tj * 16 + l16;
                    v16bf bfrag = *(const v16bf*)(W + (size_t)col * Hdim + k0 + 16 * half);
                    acc[g][tj] = __builtin_amdgcn_wmma_f32_16x16x32_bf16(
                        false, a, false, bfrag, (short)0, acc[g][tj], false, false);
                }
        }
        __syncthreads();   // barrier 1: all waves done reading h_t / x_t
        // Cell update: i/f/g/o tiles are lane/reg-aligned; c in regs; h_new -> LDS bf16
        #pragma unroll
        for (int tj = 0; tj < 2; ++tj) {
            const int j = jb + tj * 16 + l16;
            #pragma unroll
            for (int r = 0; r < 8; ++r) {
                float iv = sigm_f(acc[0][tj][r]);
                float fv = sigm_f(acc[1][tj][r]);
                float gv = tanh_fast(acc[2][tj][r]);
                float ov = sigm_f(acc[3][tj][r]);
                float cv = fv * cst[tj][r] + iv * gv;
                cst[tj][r] = cv;
                float hv = ov * tanh_fast(cv);
                h_lds[(r + 8 * half) * HSTR + j] = (__bf16)hv;
            }
        }
    };

    // ---------------- encoder: 512 steps ----------------
    load_consts(encWih, ebias);
    __syncthreads();   // h zero-init + x_0 visible

    for (int t = 0; t < Sdim; ++t) {
        do_step(encW);
        if (tid < 16) {  // stage next input (last enc step stages first dec input)
            int tn = t + 1;
            xlds[tid] = (tn < Sdim) ? src[(b0 + tid) * Sdim + tn]
                                    : tgt[(b0 + tid) * LENdim];
        }
        __syncthreads();  // barrier 2: h_{t+1}, x_{t+1} visible
    }

    // ---------------- decoder: 100 steps + fc head ----------------
    load_consts(decWih, dbias);

    for (int t = 0; t < LENdim; ++t) {
        do_step(decW);
        if (tid < 16 && (t + 1) < LENdim) xlds[tid] = tgt[(b0 + tid) * LENdim + t + 1];
        __syncthreads();  // h_{t+1} (post-update h used by fc head) visible
        if (tid < 16 * NCdim) {
            const int bi = tid / NCdim, cls = tid % NCdim;
            float s = fcb_s[cls];
            const __bf16* hr = &h_lds[bi * HSTR];
            const float*  wr = &fcw_s[cls * Hdim];
            for (int k8 = 0; k8 < Hdim / 8; ++k8) {
                v8bf hv = *(const v8bf*)(hr + k8 * 8);
                #pragma unroll
                for (int i = 0; i < 8; ++i) s += (float)hv[i] * wr[k8 * 8 + i];
            }
            out[((b0 + bi) * LENdim + t) * NCdim + cls] = s;
        }
        // pred readers rejoin at next step's internal barrier before any h write
    }
}

extern "C" void kernel_launch(void* const* d_in, const int* in_sizes, int n_in,
                              void* d_out, int out_size, void* d_ws, size_t ws_size,
                              hipStream_t stream) {
    (void)in_sizes; (void)n_in; (void)out_size; (void)ws_size;
    const float* src    = (const float*)d_in[0];
    const float* tgt    = (const float*)d_in[1];
    const float* encWih = (const float*)d_in[2];
    const float* encWhh = (const float*)d_in[3];
    const float* encBih = (const float*)d_in[4];
    const float* encBhh = (const float*)d_in[5];
    const float* decWih = (const float*)d_in[6];
    const float* decWhh = (const float*)d_in[7];
    const float* decBih = (const float*)d_in[8];
    const float* decBhh = (const float*)d_in[9];
    const float* fcW    = (const float*)d_in[10];
    const float* fcB    = (const float*)d_in[11];
    float* out = (float*)d_out;

    // Workspace layout: enc Whh bf16 (2MB) | dec Whh bf16 (2MB) | bias sums (2x8KB)
    unsigned short* encWb = (unsigned short*)d_ws;
    unsigned short* decWb = encWb + 4 * Hdim * Hdim;
    float* ebias = (float*)(decWb + 4 * Hdim * Hdim);
    float* dbias = ebias + 4 * Hdim;

    prep_kernel<<<(4 * Hdim * Hdim + 255) / 256, 256, 0, stream>>>(
        encWhh, decWhh, encWb, decWb, encBih, encBhh, decBih, decBhh, ebias, dbias);

    lstm_seq2seq_kernel<<<Bdim / 16, 512, 0, stream>>>(
        src, tgt, encWih, decWih, fcW, fcB, encWb, decWb, ebias, dbias, out);
}
